// MultiHeadAttention_23210003267868
// MI455X (gfx1250) — compile-verified, matrix operand order checked by reading, not run on hardware
//
#include <hip/hip_runtime.h>
#include <hip/hip_bf16.h>

typedef __attribute__((ext_vector_type(16))) _Float16 v16h;
typedef __attribute__((ext_vector_type(8)))  _Float16 v8h;
typedef __attribute__((ext_vector_type(2)))  _Float16 v2h;
typedef __attribute__((ext_vector_type(2)))  __fp16   v2hp;  // builtin return type
typedef __attribute__((ext_vector_type(8)))  float    v8f;

// D = A(16x32 f16) * B(32x16 f16) + C(16x16 f32), wave32
#define WMMA_F16(a, b, c) \
  __builtin_amdgcn_wmma_f32_16x16x32_f16(false, (a), false, (b), (short)0, (c), false, false)

// v_cvt_pk_f16_f32: pack two f32 -> two f16 in one VALU op
__device__ __forceinline__ v2h cvt2h(float a, float b) {
    v2hp t = __builtin_amdgcn_cvt_pkrtz(a, b);
    return __builtin_bit_cast(v2h, t);
}

// ---------------------------------------------------------------------------
// Cross-lane helpers: DPP ROW_XMASK butterfly within each group of 16 lanes
// (wave32: lanes 0-15 and 16-31 reduce independently -> matches WMMA row halves)
// ---------------------------------------------------------------------------
template <int CTRL>
__device__ __forceinline__ float dpp_movf(float x) {
    int r = __builtin_amdgcn_update_dpp(0, __builtin_bit_cast(int, x),
                                        CTRL, 0xf, 0xf, true);
    return __builtin_bit_cast(float, r);
}
__device__ __forceinline__ float rowmax16(float x) {
    x = fmaxf(x, dpp_movf<0x161>(x));  // ROW_XMASK:1
    x = fmaxf(x, dpp_movf<0x162>(x));  // ROW_XMASK:2
    x = fmaxf(x, dpp_movf<0x164>(x));  // ROW_XMASK:4
    x = fmaxf(x, dpp_movf<0x168>(x));  // ROW_XMASK:8
    return x;
}
__device__ __forceinline__ float rowsum16(float x) {
    x += dpp_movf<0x161>(x);
    x += dpp_movf<0x162>(x);
    x += dpp_movf<0x164>(x);
    x += dpp_movf<0x168>(x);
    return x;
}

// Build a 16x32 A-operand (or 32x16 B-operand) fragment for one lane.
// ISA layout (CDNA5 ISA): half 0 holds K {0..7,16..23}, half 1 holds
// K {8..15,24..31}. Row-major [m_or_n][k] tile in LDS -> two 16B ds loads.
__device__ __forceinline__ v16h lds_frag(const _Float16* rowBase, int laneHalf) {
    const v8h* p0 = reinterpret_cast<const v8h*>(rowBase + 8 * laneHalf);
    const v8h* p1 = reinterpret_cast<const v8h*>(rowBase + 8 * laneHalf + 16);
    v8h lo = *p0;
    v8h hi = *p1;
    v16h r;
#pragma unroll
    for (int i = 0; i < 8; ++i) { r[i] = lo[i]; r[i + 8] = hi[i]; }
    return r;
}

// ---------------------------------------------------------------------------
// Tiled GEMM: C[M x Nn] = A[M x K] * W[K x Nn] + bias[Nn]
// Block tile 128x128, K-step 32. 8 waves; each wave computes 64x32 (4x2 tiles).
// ---------------------------------------------------------------------------
template <bool A_F16, bool OUT_F16>
__global__ __launch_bounds__(256)
void gemm_bias_wmma(const void* __restrict__ Av, const float* __restrict__ W,
                    const float* __restrict__ bias, void* __restrict__ Cv,
                    int M, int Nn, int K) {
    constexpr int BM = 128, BN = 128, BK = 32;
    constexpr int AP = 40;  // f16 row pitch: 80B (16B aligned + bank skew)
    constexpr int BP = 40;

    __shared__ _Float16 As[BM * AP];
    __shared__ _Float16 Bs[BN * BP];  // W tile transposed: [n][k]

    const int tid  = threadIdx.x;
    const int wave = tid >> 5;
    const int lane = tid & 31;
    const int lm   = lane & 15;
    const int lh   = lane >> 4;

    const int m0 = blockIdx.y * BM;
    const int n0 = blockIdx.x * BN;
    const int wm = (wave & 1) * 64;
    const int wn = (wave >> 1) * 32;

    v8f acc[4][2];
#pragma unroll
    for (int mt = 0; mt < 4; ++mt)
#pragma unroll
        for (int nt = 0; nt < 2; ++nt) acc[mt][nt] = (v8f)0.0f;

    const int ar = tid >> 1;        // A row
    const int ac = (tid & 1) * 16;  // A k-chunk
    const int bk = tid >> 3;        // W k row
    const int bn = (tid & 7) * 16;  // W n-chunk

    for (int k0 = 0; k0 < K; k0 += BK) {
        // ---- stage A tile [128 x 32] as f16 row-major [m][k] ----
        if (A_F16) {
            const _Float16* g = (const _Float16*)Av + (size_t)(m0 + ar) * K + k0 + ac;
            v8h a0 = *(const v8h*)g;
            v8h a1 = *(const v8h*)(g + 8);
            *(v8h*)&As[ar * AP + ac]     = a0;
            *(v8h*)&As[ar * AP + ac + 8] = a1;
        } else {
            const float* g = (const float*)Av + (size_t)(m0 + ar) * K + k0 + ac;
            v8h p0, p1;
#pragma unroll
            for (int i = 0; i < 8; i += 4) {
                float4 f = *(const float4*)(g + i);
                v2h h01 = cvt2h(f.x, f.y);
                v2h h23 = cvt2h(f.z, f.w);
                p0[i] = h01[0]; p0[i+1] = h01[1]; p0[i+2] = h23[0]; p0[i+3] = h23[1];
            }
#pragma unroll
            for (int i = 0; i < 8; i += 4) {
                float4 f = *(const float4*)(g + 8 + i);
                v2h h01 = cvt2h(f.x, f.y);
                v2h h23 = cvt2h(f.z, f.w);
                p1[i] = h01[0]; p1[i+1] = h01[1]; p1[i+2] = h23[0]; p1[i+3] = h23[1];
            }
            *(v8h*)&As[ar * AP + ac]     = p0;
            *(v8h*)&As[ar * AP + ac + 8] = p1;
        }
        if (k0 + BK < K) {
            const char* base = (const char*)Av;
            size_t esz = A_F16 ? 2u : 4u;
            __builtin_prefetch(base + ((size_t)(m0 + ar) * K + k0 + BK + ac) * esz, 0, 1);
        }

        // ---- stage W tile [32 x 128] transposed into LDS as [n][k] ----
        {
            const float* g = W + (size_t)(k0 + bk) * Nn + n0 + bn;
            float wv[16];
#pragma unroll
            for (int i = 0; i < 16; i += 4) {
                float4 f = *(const float4*)(g + i);
                wv[i] = f.x; wv[i+1] = f.y; wv[i+2] = f.z; wv[i+3] = f.w;
            }
#pragma unroll
            for (int i = 0; i < 16; i += 2) {
                v2h hh = cvt2h(wv[i], wv[i+1]);
                Bs[(bn + i)     * BP + bk] = hh[0];
                Bs[(bn + i + 1) * BP + bk] = hh[1];
            }
        }
        __syncthreads();

        // ---- compute: 4x2 wmma tiles per wave ----
        v16h af[4], bf[2];
#pragma unroll
        for (int mt = 0; mt < 4; ++mt)
            af[mt] = lds_frag(&As[(wm + mt * 16 + lm) * AP], lh);
#pragma unroll
        for (int nt = 0; nt < 2; ++nt)
            bf[nt] = lds_frag(&Bs[(wn + nt * 16 + lm) * BP], lh);
#pragma unroll
        for (int mt = 0; mt < 4; ++mt)
#pragma unroll
            for (int nt = 0; nt < 2; ++nt)
                acc[mt][nt] = WMMA_F16(af[mt], bf[nt], acc[mt][nt]);
        __syncthreads();
    }

    // ---- epilogue: bias add, store (lane holds col lm, rows r+8*lh) ----
#pragma unroll
    for (int nt = 0; nt < 2; ++nt) {
        const int col = n0 + wn + nt * 16 + lm;
        const float bv = bias[col];
#pragma unroll
        for (int mt = 0; mt < 4; ++mt) {
            const int rowb = m0 + wm + mt * 16 + 8 * lh;
            if (OUT_F16) {
#pragma unroll
                for (int r = 0; r < 8; r += 2) {
                    v2h hh = cvt2h(acc[mt][nt][r] + bv, acc[mt][nt][r+1] + bv);
                    ((_Float16*)Cv)[(size_t)(rowb + r)     * Nn + col] = hh[0];
                    ((_Float16*)Cv)[(size_t)(rowb + r + 1) * Nn + col] = hh[1];
                }
            } else {
#pragma unroll
                for (int r = 0; r < 8; ++r)
                    ((float*)Cv)[(size_t)(rowb + r) * Nn + col] = acc[mt][nt][r] + bv;
            }
        }
    }
}

// ---------------------------------------------------------------------------
// Flash attention. qkv: f16 [B*N][3*D]; q|k|v each D=1024 = 16 heads x 64.
// Grid: (N/128, H, B). 8 waves; wave w owns 16 q rows. 64-key chunks.
// ---------------------------------------------------------------------------
__global__ __launch_bounds__(256)
void attention_wmma(const _Float16* __restrict__ qkv, _Float16* __restrict__ out) {
    constexpr int SEQ = 1024, HD = 64, D3 = 3072;
    constexpr int QP = 72;  // f16 row pitch: 144B
    // softmax: exp((s-m)/sqrt(64)) = 2^((s-m)*log2e/8)
    constexpr float ESC = 0.18033688011112042592f;

    __shared__ _Float16 Qs[128 * QP];
    __shared__ _Float16 Ks[64 * QP];      // [key_row][hd]  (B operand, Q*K^T)
    __shared__ _Float16 Vs[64 * QP];      // [hd][key_row]  (B operand, P*V)
    __shared__ _Float16 Ps[8 * 16 * QP];  // per-wave P scratch [row][key]

    const int tid  = threadIdx.x;
    const int wave = tid >> 5;
    const int lane = tid & 31;
    const int lm   = lane & 15;
    const int lh   = lane >> 4;

    const int q0 = blockIdx.x * 128;
    const int h  = blockIdx.y;
    const int b  = blockIdx.z;

    const _Float16* qBase = qkv + (size_t)b * SEQ * D3 + h * HD;

    // ---- load Q tile: 128 rows x 64 f16 ----
    {
        const int r  = tid >> 1;
        const int c0 = (tid & 1) * 32;
        const _Float16* g = qBase + (size_t)(q0 + r) * D3 + c0;
#pragma unroll
        for (int i = 0; i < 4; ++i)
            *(v8h*)&Qs[r * QP + c0 + 8 * i] = *(const v8h*)(g + 8 * i);
    }
    __syncthreads();

    // loop-invariant Q fragments (k = 0..31 and 32..63 of head dim)
    const v16h aq0 = lds_frag(&Qs[(wave * 16 + lm) * QP], lh);
    const v16h aq1 = lds_frag(&Qs[(wave * 16 + lm) * QP + 32], lh);

    // per-lane state: rows r+8*lh; mrow lane-uniform per half, lrow is a
    // per-lane PARTIAL sum (reduced across the 16 lanes only at the end)
    float mrow[8], lrow[8];
    v8f oacc[4];
#pragma unroll
    for (int r = 0; r < 8; ++r) { mrow[r] = -1e30f; lrow[r] = 0.0f; }
#pragma unroll
    for (int ht = 0; ht < 4; ++ht) oacc[ht] = (v8f)0.0f;

    _Float16* pw = &Ps[wave * 16 * QP];

    for (int kb = 0; kb < SEQ / 64; ++kb) {
        __syncthreads();  // previous K/V chunk fully consumed
        {
            const int kr = tid >> 2;
            const int c0 = (tid & 3) * 16;
            const _Float16* gk = qBase + (size_t)(kb * 64 + kr) * D3 + 1024 + c0;
            *(v8h*)&Ks[kr * QP + c0]     = *(const v8h*)gk;
            *(v8h*)&Ks[kr * QP + c0 + 8] = *(const v8h*)(gk + 8);

            const _Float16* gv = qBase + (size_t)(kb * 64 + kr) * D3 + 2048 + c0;
            v8h v0 = *(const v8h*)gv;
            v8h v1 = *(const v8h*)(gv + 8);
#pragma unroll
            for (int i = 0; i < 8; ++i) {           // transpose V into [hd][key]
                Vs[(c0 + i)     * QP + kr] = v0[i];
                Vs[(c0 + 8 + i) * QP + kr] = v1[i];
            }
        }
        __syncthreads();

        // ---- S = Q * K^T (raw scores; scale folded into exp2 constant) ----
        v8f s[4];
#pragma unroll
        for (int nt = 0; nt < 4; ++nt) {
            v8f c = (v8f)0.0f;
            v16h bk0 = lds_frag(&Ks[(nt * 16 + lm) * QP], lh);
            c = WMMA_F16(aq0, bk0, c);
            v16h bk1 = lds_frag(&Ks[(nt * 16 + lm) * QP + 32], lh);
            c = WMMA_F16(aq1, bk1, c);
            s[nt] = c;
        }

        // ---- online softmax: DPP row-max, deferred row-sum ----
#pragma unroll
        for (int r = 0; r < 8; ++r) {
            float mx = fmaxf(fmaxf(s[0][r], s[1][r]), fmaxf(s[2][r], s[3][r]));
            mx = rowmax16(mx);
            const float mnew  = fmaxf(mrow[r], mx);
            const float alpha = __builtin_amdgcn_exp2f((mrow[r] - mnew) * ESC);
            float psum = 0.0f;
#pragma unroll
            for (int nt = 0; nt < 4; ++nt) {
                const float p = __builtin_amdgcn_exp2f((s[nt][r] - mnew) * ESC);
                s[nt][r] = p;
                psum += p;
            }
            lrow[r] = lrow[r] * alpha + psum;  // per-lane partial
            mrow[r] = mnew;
#pragma unroll
            for (int ht = 0; ht < 4; ++ht) oacc[ht][r] *= alpha;
        }

        // ---- P -> LDS (C-layout -> A-layout; same-wave DS is in-order) ----
#pragma unroll
        for (int nt = 0; nt < 4; ++nt)
#pragma unroll
            for (int r = 0; r < 8; r += 2) {
                v2h ph = cvt2h(s[nt][r], s[nt][r + 1]);
                pw[(r     + 8 * lh) * QP + nt * 16 + lm] = ph[0];
                pw[(r + 1 + 8 * lh) * QP + nt * 16 + lm] = ph[1];
            }

        // ---- O += P * V ----
        v16h pa0 = lds_frag(&pw[lm * QP], lh);
        v16h pa1 = lds_frag(&pw[lm * QP + 32], lh);
#pragma unroll
        for (int ht = 0; ht < 4; ++ht) {
            v16h bv0 = lds_frag(&Vs[(ht * 16 + lm) * QP], lh);
            oacc[ht] = WMMA_F16(pa0, bv0, oacc[ht]);
            v16h bv1 = lds_frag(&Vs[(ht * 16 + lm) * QP + 32], lh);
            oacc[ht] = WMMA_F16(pa1, bv1, oacc[ht]);
        }
    }

    // ---- final row-sum reduce, normalize, store f16 ----
    float inv[8];
#pragma unroll
    for (int r = 0; r < 8; ++r) inv[r] = 1.0f / rowsum16(lrow[r]);

    _Float16* oBase = out + ((size_t)b * SEQ + q0 + wave * 16 + 8 * lh) * 1024 + h * HD;
#pragma unroll
    for (int ht = 0; ht < 4; ++ht) {
        const int col = ht * 16 + lm;
#pragma unroll
        for (int r = 0; r < 8; r += 2) {
            v2h oh = cvt2h(oacc[ht][r] * inv[r], oacc[ht][r + 1] * inv[r + 1]);
            oBase[(size_t)r       * 1024 + col] = oh[0];
            oBase[(size_t)(r + 1) * 1024 + col] = oh[1];
        }
    }
}

// ---------------------------------------------------------------------------
extern "C" void kernel_launch(void* const* d_in, const int* in_sizes, int n_in,
                              void* d_out, int out_size, void* d_ws, size_t ws_size,
                              hipStream_t stream) {
    const float* x     = (const float*)d_in[0];  // [16,1024,1024]
    const float* w_qkv = (const float*)d_in[1];  // [1024,3072]
    const float* b_qkv = (const float*)d_in[2];  // [3072]
    const float* w_out = (const float*)d_in[3];  // [1024,1024]
    const float* b_out = (const float*)d_in[4];  // [1024]
    float* out = (float*)d_out;                  // [16,1024,1024] fp32

    constexpr int M = 16 * 1024;  // B*N
    constexpr int D = 1024;

    _Float16* qkv_h  = (_Float16*)d_ws;                                   // 96 MB
    _Float16* attn_h = (_Float16*)((char*)d_ws + (size_t)M * 3 * D * 2);  // 32 MB

    dim3 g1(3 * D / 128, M / 128);
    gemm_bias_wmma<false, true><<<g1, 256, 0, stream>>>(x, w_qkv, b_qkv, qkv_h,
                                                        M, 3 * D, D);
    dim3 g2(1024 / 128, 16, 16);
    attention_wmma<<<g2, 256, 0, stream>>>(qkv_h, attn_h);

    dim3 g3(D / 128, M / 128);
    gemm_bias_wmma<true, false><<<g3, 256, 0, stream>>>(attn_h, w_out, b_out, out,
                                                        M, D, D);
}
